// Attention_31353261260890
// MI455X (gfx1250) — matrix operand order repaired and checked by reading, not compiled, not run on hardware
//
#include <hip/hip_runtime.h>
#include <stdint.h>

#define DMODEL 1024
#define SEQ    2048
#define NBATCH 4
#define NHEAD  16
#define HDIM   64
#define MTOT   (NBATCH*SEQ)                    // 8192
#define ELEMS_PER_BUF (NBATCH*NHEAD*SEQ*HDIM)  // 8388608
#define LOG2E  1.4426950408889634f

typedef __attribute__((ext_vector_type(16))) __bf16    v16bf;
typedef __attribute__((ext_vector_type(2)))  __bf16    bf16x2;
typedef __attribute__((ext_vector_type(8)))  float     v8f;
typedef __attribute__((ext_vector_type(4)))  float     f32x4;
typedef __attribute__((ext_vector_type(4)))  uint32_t  u32x4;

union Frag16 { v16bf v; uint32_t u[8]; u32x4 q[2]; };

__device__ __forceinline__ uint32_t pk_bf16(float lo, float hi) {
#if __has_builtin(__builtin_amdgcn_cvt_pk_bf16_f32)
    union { bf16x2 h; uint32_t u; } cv;
    cv.h = __builtin_amdgcn_cvt_pk_bf16_f32(lo, hi);
    return cv.u;
#else
    uint32_t a = __builtin_bit_cast(uint32_t, lo);
    uint32_t b = __builtin_bit_cast(uint32_t, hi);
    a = (a + 0x7FFFu + ((a >> 16) & 1u)) >> 16;
    b = (b + 0x7FFFu + ((b >> 16) & 1u)) & 0xFFFF0000u;
    return a | b;
#endif
}
__device__ __forceinline__ unsigned short f2bf(float f) {
    return (unsigned short)(pk_bf16(f, f) & 0xFFFFu);
}
__device__ __forceinline__ float fast_exp2(float x) {
#if __has_builtin(__builtin_amdgcn_exp2f)
    return __builtin_amdgcn_exp2f(x);
#else
    return exp2f(x);
#endif
}
__device__ __forceinline__ float fast_rcp(float x) {
#if __has_builtin(__builtin_amdgcn_rcpf)
    return __builtin_amdgcn_rcpf(x);
#else
    return 1.f / x;
#endif
}
__device__ __forceinline__ v8f wmma_bf(const Frag16& a, const Frag16& b, v8f c) {
    return __builtin_amdgcn_wmma_f32_16x16x32_bf16(false, a.v, false, b.v,
                                                   (short)0, c, false, false);
}

// ------------------- CDNA5 async global->LDS copy (guarded) -----------------
#if __has_builtin(__builtin_amdgcn_global_load_async_to_lds_b128) && \
    __has_builtin(__builtin_amdgcn_s_wait_asynccnt)
#define HAVE_ASYNC_LDS 1
#endif

typedef int v4i_vs __attribute__((vector_size(16)));
typedef __attribute__((address_space(1))) v4i_vs* g_v4i;
typedef __attribute__((address_space(3))) v4i_vs* l_v4i;

__device__ __forceinline__ void cp16_g2l(const void* g, void* l) {
#ifdef HAVE_ASYNC_LDS
    __builtin_amdgcn_global_load_async_to_lds_b128((g_v4i)(void*)g, (l_v4i)l, 0, 0);
#else
    *(u32x4*)l = *(const u32x4*)g;
#endif
}
__device__ __forceinline__ void async_wait4() {
#ifdef HAVE_ASYNC_LDS
    __builtin_amdgcn_s_wait_asynccnt(4);
#endif
}
__device__ __forceinline__ void async_wait0() {
#ifdef HAVE_ASYNC_LDS
    __builtin_amdgcn_s_wait_asynccnt(0);
#endif
}

// ---------------------------------------------------------------------------
// Kernel 0: fp32 -> bf16 conversion pass (bandwidth-bound, runs once).
// ---------------------------------------------------------------------------
__global__ __launch_bounds__(256) void k_cvt(const float* __restrict__ src,
                                             unsigned short* __restrict__ dst,
                                             int n8)
{
    int i = blockIdx.x * 256 + threadIdx.x;
    if (i >= n8) return;
    const f32x4* p = (const f32x4*)src + 2 * (size_t)i;
    f32x4 a = p[0];
    f32x4 b = p[1];
    uint32_t o[4] = { pk_bf16(a.x, a.y), pk_bf16(a.z, a.w),
                      pk_bf16(b.x, b.y), pk_bf16(b.z, b.w) };
    *((u32x4*)dst + i) = *(u32x4*)o;
}

// ---------------------------------------------------------------------------
// Kernel 1: QKV projection (bf16 in, f32 accum), per-wave 64x64 tile.
// Q,K -> [B,H,S,64] bf16 ; V -> [B,H,64,S] bf16 (transposed for PV matmul)
// ---------------------------------------------------------------------------
__global__ __launch_bounds__(32) void k_qkv(
    const unsigned short* __restrict__ Xb,
    const unsigned short* __restrict__ Wq, const float* __restrict__ bq,
    const unsigned short* __restrict__ Wk, const float* __restrict__ bk,
    const unsigned short* __restrict__ Wv, const float* __restrict__ bv,
    unsigned short* __restrict__ Q, unsigned short* __restrict__ K,
    unsigned short* __restrict__ Vt)
{
    const int lane = threadIdx.x & 31;
    const int g = lane >> 4, ln = lane & 15;
    const int n0 = blockIdx.x * 64;
    const int m0 = blockIdx.y * 64;
    const int which = blockIdx.z;
    const unsigned short* w = (which == 0) ? Wq : (which == 1) ? Wk : Wv;
    const float* bs          = (which == 0) ? bq : (which == 1) ? bk : bv;

    v8f acc[4][4];
    for (int nt = 0; nt < 4; ++nt) {
        float bval = bs[n0 + nt * 16 + ln];
        for (int mt = 0; mt < 4; ++mt)
            for (int r = 0; r < 8; ++r) acc[mt][nt][r] = bval;
    }
    const unsigned short* arow[4];
    for (int mt = 0; mt < 4; ++mt)
        arow[mt] = Xb + (size_t)(m0 + mt * 16 + ln) * DMODEL;
    const unsigned short* brow[4];
    for (int nt = 0; nt < 4; ++nt)
        brow[nt] = w + (size_t)(n0 + nt * 16 + ln) * DMODEL;

    for (int k0 = 0; k0 < DMODEL; k0 += 32) {
        if (k0 + 256 < DMODEL) {
            __builtin_prefetch(arow[0] + k0 + 256, 0, 3);
            __builtin_prefetch(arow[2] + k0 + 256, 0, 3);
        }
        Frag16 a[4], b[4];
        for (int mt = 0; mt < 4; ++mt) {
            a[mt].q[0] = *(const u32x4*)(arow[mt] + k0 + g * 8);
            a[mt].q[1] = *(const u32x4*)(arow[mt] + k0 + 16 + g * 8);
        }
        for (int nt = 0; nt < 4; ++nt) {
            b[nt].q[0] = *(const u32x4*)(brow[nt] + k0 + g * 8);
            b[nt].q[1] = *(const u32x4*)(brow[nt] + k0 + 16 + g * 8);
        }
        for (int mt = 0; mt < 4; ++mt)
            for (int nt = 0; nt < 4; ++nt)
                acc[mt][nt] = wmma_bf(a[mt], b[nt], acc[mt][nt]);
    }

    if (which < 2) {
        unsigned short* dst = (which == 0) ? Q : K;
        for (int nt = 0; nt < 4; ++nt) {
            int ncol = n0 + nt * 16 + ln;
            int h = ncol >> 6, d = ncol & 63;
            for (int mt = 0; mt < 4; ++mt) {
                for (int r = 0; r < 8; ++r) {
                    int m = m0 + mt * 16 + g * 8 + r;
                    int bb = m >> 11, s = m & (SEQ - 1);
                    dst[(((size_t)(bb * NHEAD + h) * SEQ + s) << 6) + d] =
                        f2bf(acc[mt][nt][r]);
                }
            }
        }
    } else {
        for (int nt = 0; nt < 4; ++nt) {
            int ncol = n0 + nt * 16 + ln;
            int h = ncol >> 6, d = ncol & 63;
            for (int mt = 0; mt < 4; ++mt) {
                int mb = m0 + mt * 16 + g * 8;
                int bb = mb >> 11, s = mb & (SEQ - 1);
                uint32_t pk[4];
                for (int i = 0; i < 4; ++i)
                    pk[i] = pk_bf16(acc[mt][nt][2 * i], acc[mt][nt][2 * i + 1]);
                *(u32x4*)(Vt + (((size_t)(bb * NHEAD + h) * HDIM + d) << 11) + s) =
                    *(u32x4*)pk;
            }
        }
    }
}

// ---------------------------------------------------------------------------
// Kernel 2: flash attention. Block = 4 waves on one (b,h); K/V tiles for each
// 32-key block staged into LDS once per block via async global->LDS copies,
// double buffered so the fill overlaps WMMA+softmax of the current block.
// ---------------------------------------------------------------------------
__global__ __launch_bounds__(128) void k_attn(
    const unsigned short* __restrict__ Q, const unsigned short* __restrict__ K,
    const unsigned short* __restrict__ Vt, unsigned short* __restrict__ O)
{
    __shared__ __align__(16) unsigned short Klds[2][32][72];  // 32 keys x 64 d (+pad)
    __shared__ __align__(16) unsigned short Vlds[2][64][40];  // 64 d x 32 keys (+pad)
    __shared__ float Pld[4][16][33];

    const int tid = threadIdx.x;
    const int wv_ = tid >> 5, lane = tid & 31;
    const int g = lane >> 4, ln = lane & 15;
    const int bb = blockIdx.z, h = blockIdx.y;
    const int s0 = (blockIdx.x * 4 + wv_) * 16;
    const float SCL = 0.125f * LOG2E;   // 1/sqrt(64) folded into exp2 domain

    const unsigned short* Qp = Q + (size_t)(bb * NHEAD + h) * SEQ * HDIM;
    const unsigned short* Kp = K + (size_t)(bb * NHEAD + h) * SEQ * HDIM;
    const unsigned short* Vp = Vt + (size_t)(bb * NHEAD + h) * HDIM * SEQ;

    Frag16 qf[2];
    {
        const unsigned short* qr = Qp + (size_t)(s0 + ln) * HDIM;
        for (int kq = 0; kq < 2; ++kq) {
            qf[kq].q[0] = *(const u32x4*)(qr + kq * 32 + g * 8);
            qf[kq].q[1] = *(const u32x4*)(qr + kq * 32 + 16 + g * 8);
        }
    }

    // Cooperative fill of one 32-key K/V tile pair (4 async b128 per wave).
    auto fill = [&](int kb) {
        const int buf = kb & 1;
        const int key0 = kb * 32;
        for (int i = 0; i < 2; ++i) {               // K: 256 x 16B chunks
            int c = tid + i * 128;
            int row = c >> 3, col = (c & 7) * 8;
            cp16_g2l(Kp + (size_t)(key0 + row) * HDIM + col, &Klds[buf][row][col]);
        }
        for (int i = 0; i < 2; ++i) {               // V^T: 256 x 16B chunks
            int c = tid + i * 128;
            int d = c >> 2, col = (c & 3) * 8;
            cp16_g2l(Vp + (size_t)d * SEQ + key0 + col, &Vlds[buf][d][col]);
        }
    };

    float mrow[8], lrow[8];
    v8f oacc[4];
    for (int r = 0; r < 8; ++r) { mrow[r] = -__builtin_inff(); lrow[r] = 0.f; }
    for (int nt = 0; nt < 4; ++nt)
        for (int r = 0; r < 8; ++r) oacc[nt][r] = 0.f;

    fill(0);
    fill(1);

    for (int kb = 0; kb < SEQ / 32; ++kb) {
        const int buf = kb & 1;
        if (kb < SEQ / 32 - 1) async_wait4();   // this block's fill done, next may fly
        else                   async_wait0();
        __syncthreads();

        v8f sc[2];
        for (int t = 0; t < 2; ++t) {
            const unsigned short* kr = &Klds[buf][t * 16 + ln][0];
            Frag16 kf0, kf1;
            kf0.q[0] = *(const u32x4*)(kr + g * 8);
            kf0.q[1] = *(const u32x4*)(kr + 16 + g * 8);
            kf1.q[0] = *(const u32x4*)(kr + 32 + g * 8);
            kf1.q[1] = *(const u32x4*)(kr + 48 + g * 8);
            v8f z;
            for (int r = 0; r < 8; ++r) z[r] = 0.f;
            z = wmma_bf(qf[0], kf0, z);
            z = wmma_bf(qf[1], kf1, z);
            sc[t] = z;
        }
        for (int t = 0; t < 2; ++t)
            for (int r = 0; r < 8; ++r) sc[t][r] *= SCL;

        for (int r = 0; r < 8; ++r) {
            float vmax = fmaxf(sc[0][r], sc[1][r]);
            for (int off = 1; off < 16; off <<= 1)
                vmax = fmaxf(vmax, __shfl_xor(vmax, off, 32));
            float newm = fmaxf(mrow[r], vmax);
            float alpha = fast_exp2(mrow[r] - newm);
            float e0 = fast_exp2(sc[0][r] - newm);
            float e1 = fast_exp2(sc[1][r] - newm);
            float psum = e0 + e1;
            for (int off = 1; off < 16; off <<= 1)
                psum += __shfl_xor(psum, off, 32);
            lrow[r] = lrow[r] * alpha + psum;
            mrow[r] = newm;
            for (int nt = 0; nt < 4; ++nt) oacc[nt][r] *= alpha;
            int row = r + g * 8;
            Pld[wv_][row][ln] = e0;
            Pld[wv_][row][16 + ln] = e1;
        }
        asm volatile("s_wait_dscnt 0" ::: "memory");

        Frag16 pf;
        {
            const float* pr = &Pld[wv_][ln][0];
            for (int i = 0; i < 4; ++i)
                pf.u[i] = pk_bf16(pr[g * 8 + 2 * i], pr[g * 8 + 2 * i + 1]);
            for (int i = 0; i < 4; ++i)
                pf.u[4 + i] = pk_bf16(pr[16 + g * 8 + 2 * i], pr[16 + g * 8 + 2 * i + 1]);
        }
        for (int nt = 0; nt < 4; ++nt) {
            const unsigned short* vr = &Vlds[buf][nt * 16 + ln][g * 8];
            Frag16 vf;
            vf.q[0] = *(const u32x4*)(vr);
            vf.q[1] = *(const u32x4*)(vr + 16);
            oacc[nt] = wmma_bf(pf, vf, oacc[nt]);
        }

        __syncthreads();                      // all waves done reading buf
        if (kb + 2 < SEQ / 32) fill(kb + 2);  // overwrite it for block kb+2
    }

    for (int r = 0; r < 8; ++r) {
        float inv = fast_rcp(lrow[r]);
        for (int nt = 0; nt < 4; ++nt) oacc[nt][r] *= inv;
    }
    unsigned short* Op = O + (size_t)(bb * NHEAD + h) * SEQ * HDIM;
    for (int nt = 0; nt < 4; ++nt) {
        int d = nt * 16 + ln;
        for (int r = 0; r < 8; ++r) {
            int s = s0 + g * 8 + r;
            Op[(size_t)s * HDIM + d] = f2bf(oacc[nt][r]);
        }
    }
}

// ---------------------------------------------------------------------------
// Kernel 3: output projection.  out[m,n] = sum_k O[m,k]*Wo[n,k] + bo[n]
// O is bf16 in [B,H,S,64]; k -> (h = k/64, d = k%64).  Per-wave 64x64 tile.
// ---------------------------------------------------------------------------
__global__ __launch_bounds__(32) void k_oproj(
    const unsigned short* __restrict__ Ob,
    const unsigned short* __restrict__ Wo, const float* __restrict__ bo,
    float* __restrict__ out)
{
    const int lane = threadIdx.x & 31;
    const int g = lane >> 4, ln = lane & 15;
    const int n0 = blockIdx.x * 64;
    const int m0 = blockIdx.y * 64;

    v8f acc[4][4];
    for (int nt = 0; nt < 4; ++nt) {
        float bval = bo[n0 + nt * 16 + ln];
        for (int mt = 0; mt < 4; ++mt)
            for (int r = 0; r < 8; ++r) acc[mt][nt][r] = bval;
    }
    const unsigned short* abase[4];
    for (int mt = 0; mt < 4; ++mt) {
        int m = m0 + mt * 16 + ln;
        int bb = m >> 11, s = m & (SEQ - 1);
        abase[mt] = Ob + (size_t)(bb * NHEAD) * SEQ * HDIM + (size_t)s * HDIM;
    }
    const unsigned short* brow[4];
    for (int nt = 0; nt < 4; ++nt)
        brow[nt] = Wo + (size_t)(n0 + nt * 16 + ln) * DMODEL;

    for (int k0 = 0; k0 < DMODEL; k0 += 32) {
        int hh = k0 >> 6;
        int dbase = k0 & 63;
        Frag16 a[4], b[4];
        for (int mt = 0; mt < 4; ++mt) {
            const unsigned short* p = abase[mt] + (size_t)hh * SEQ * HDIM + dbase;
            a[mt].q[0] = *(const u32x4*)(p + g * 8);
            a[mt].q[1] = *(const u32x4*)(p + 16 + g * 8);
        }
        for (int nt = 0; nt < 4; ++nt) {
            b[nt].q[0] = *(const u32x4*)(brow[nt] + k0 + g * 8);
            b[nt].q[1] = *(const u32x4*)(brow[nt] + k0 + 16 + g * 8);
        }
        for (int mt = 0; mt < 4; ++mt)
            for (int nt = 0; nt < 4; ++nt)
                acc[mt][nt] = wmma_bf(a[mt], b[nt], acc[mt][nt]);
    }

    for (int nt = 0; nt < 4; ++nt) {
        int ncol = n0 + nt * 16 + ln;
        for (int mt = 0; mt < 4; ++mt)
            for (int r = 0; r < 8; ++r) {
                int m = m0 + mt * 16 + g * 8 + r;
                out[(size_t)m * DMODEL + ncol] = acc[mt][nt][r];
            }
    }
}

// ---------------------------------------------------------------------------
extern "C" void kernel_launch(void* const* d_in, const int* in_sizes, int n_in,
                              void* d_out, int out_size, void* d_ws, size_t ws_size,
                              hipStream_t stream) {
    const float* x    = (const float*)d_in[0];
    const float* wq_w = (const float*)d_in[1];
    const float* wq_b = (const float*)d_in[2];
    const float* wk_w = (const float*)d_in[3];
    const float* wk_b = (const float*)d_in[4];
    const float* wv_w = (const float*)d_in[5];
    const float* wv_b = (const float*)d_in[6];
    const float* wo_w = (const float*)d_in[7];
    const float* wo_b = (const float*)d_in[8];

    unsigned short* Q   = (unsigned short*)d_ws;
    unsigned short* K   = Q + (size_t)ELEMS_PER_BUF;
    unsigned short* Vt  = K + (size_t)ELEMS_PER_BUF;
    unsigned short* O   = Vt + (size_t)ELEMS_PER_BUF;
    unsigned short* Xb  = O + (size_t)ELEMS_PER_BUF;
    unsigned short* Wqb = Xb + (size_t)MTOT * DMODEL;
    unsigned short* Wkb = Wqb + (size_t)DMODEL * DMODEL;
    unsigned short* Wvb = Wkb + (size_t)DMODEL * DMODEL;
    unsigned short* Wob = Wvb + (size_t)DMODEL * DMODEL;

    {
        int n8x = (MTOT * DMODEL) / 8;
        int n8w = (DMODEL * DMODEL) / 8;
        k_cvt<<<dim3((n8x + 255) / 256), 256, 0, stream>>>(x, Xb, n8x);
        k_cvt<<<dim3((n8w + 255) / 256), 256, 0, stream>>>(wq_w, Wqb, n8w);
        k_cvt<<<dim3((n8w + 255) / 256), 256, 0, stream>>>(wk_w, Wkb, n8w);
        k_cvt<<<dim3((n8w + 255) / 256), 256, 0, stream>>>(wv_w, Wvb, n8w);
        k_cvt<<<dim3((n8w + 255) / 256), 256, 0, stream>>>(wo_w, Wob, n8w);
    }

    dim3 gq(DMODEL / 64, MTOT / 64, 3);
    k_qkv<<<gq, 32, 0, stream>>>(Xb, Wqb, wq_b, Wkb, wk_b, Wvb, wv_b, Q, K, Vt);

    dim3 ga(SEQ / 64, NHEAD, NBATCH);
    k_attn<<<ga, 128, 0, stream>>>(Q, K, Vt, O);

    dim3 go(DMODEL / 64, MTOT / 64, 1);
    k_oproj<<<go, 32, 0, stream>>>(O, Wob, wo_b, (float*)d_out);
}
